// ConvolutinalAttentionMLP_70308614635906
// MI455X (gfx1250) — matrix operand order repaired and checked by reading, not compiled, not run on hardware
//
#include <hip/hip_runtime.h>
#include <hip/hip_bf16.h>
#include <math.h>

// ---------------------------------------------------------------------------
// Model dims (compile-time constants from the reference)
// ---------------------------------------------------------------------------
#define BATCH 256
#define SEQ   4
#define HID   2048
#define ROWS  (BATCH * SEQ)        // 1024 token rows
#define EMBED_IN 3072
#define NCLASS 1000
#define ATTN_SCALE 0.022097086912079608f   // 1/sqrt(2048)

// GEMM tiling: 128x64 block tile, 8 wave32s each owning 16 rows x 64 cols
#define BM 128
#define BN 64
#define BK 32

typedef __bf16 bf16_t;
typedef bf16_t v16bf __attribute__((ext_vector_type(16)));
typedef float  v8f   __attribute__((ext_vector_type(8)));
typedef int    v4i_g __attribute__((vector_size(16)));   // matches builtin's V4i

// Async global->LDS path (guarded; falls back to sync loads if unavailable)
#define USE_ASYNC_LDS 0
#if defined(__has_builtin)
#if __has_builtin(__builtin_amdgcn_global_load_async_to_lds_b128) && \
    __has_builtin(__builtin_amdgcn_s_wait_asynccnt)
#undef USE_ASYNC_LDS
#define USE_ASYNC_LDS 1
#endif
#endif

__device__ __forceinline__ v16bf pack16(float4 a, float4 b, float4 c, float4 d) {
  v16bf v;
  v[0]  = (bf16_t)a.x; v[1]  = (bf16_t)a.y; v[2]  = (bf16_t)a.z; v[3]  = (bf16_t)a.w;
  v[4]  = (bf16_t)b.x; v[5]  = (bf16_t)b.y; v[6]  = (bf16_t)b.z; v[7]  = (bf16_t)b.w;
  v[8]  = (bf16_t)c.x; v[9]  = (bf16_t)c.y; v[10] = (bf16_t)c.z; v[11] = (bf16_t)c.w;
  v[12] = (bf16_t)d.x; v[13] = (bf16_t)d.y; v[14] = (bf16_t)d.z; v[15] = (bf16_t)d.w;
  return v;
}

// ---------------------------------------------------------------------------
// 1) Patchify + mask:  x(B,3,64,64) -> t(ROWS, EMBED_IN), fused patch_mask mul
// ---------------------------------------------------------------------------
__global__ __launch_bounds__(256) void patch_kernel(
    const float* __restrict__ x, const float* __restrict__ mask,
    float* __restrict__ t) {
  int idx = blockIdx.x * 256 + threadIdx.x;
  const int total = BATCH * 12288;                 // 12288 = N*D = SEQ*EMBED_IN
  if (idx >= total) return;
  int b = idx / 12288, f = idx % 12288;
  int n = f / 48, d = f % 48;
  int c = d >> 4, rem = d & 15, i = rem >> 2, j = rem & 3;
  int pr = n >> 4, pc = n & 15;
  float v = x[(((size_t)b * 3 + c) * 64 + pr * 4 + i) * 64 + pc * 4 + j];
  t[idx] = v * mask[idx];
}

// ---------------------------------------------------------------------------
// 2) Generic bf16-WMMA GEMM:  C = epilogue(A[MxK] @ W[KxN] + bias) + rs*R
//    act: 0 = none, 1 = exact GELU, 2 = leaky_relu(0.2)
//    Double-buffered LDS: A tile fp32 via async DMA, B tile bf16 (converted
//    once at staging so fragment builds are conversion-free 32B LDS loads).
// ---------------------------------------------------------------------------
__global__ __launch_bounds__(256) void gemm_kernel(
    const float* __restrict__ A, const float* __restrict__ W,
    const float* __restrict__ bias, const float* __restrict__ R,
    float resScale, float* __restrict__ C,
    int M, int N, int K, int act) {
  __shared__ float  As[2][BM * BK];     // As[buf][row][k], fp32 (async DMA dest)
  __shared__ bf16_t Bs[2][BN * BK];     // Bs[buf][col][k], bf16 (transposed)

  const int tid  = threadIdx.x;
  const int wr   = tid >> 5;                   // wave id = row strip (0..7)
  const int lane = tid & 31;
  const int half = lane >> 4, r16 = lane & 15;
  const int m0 = blockIdx.y * BM;
  const int n0 = blockIdx.x * BN;
  const bool fullN = (n0 + BN <= N);

  v8f acc[4];
  #pragma unroll
  for (int ct = 0; ct < 4; ++ct)
    acc[ct] = v8f{0.f, 0.f, 0.f, 0.f, 0.f, 0.f, 0.f, 0.f};

  // staging assignments
  const int ar = tid >> 1, ak = (tid & 1) * 16; // A sync path: row ar, 16 floats
  const int bk = tid >> 3, bn = (tid & 7) * 8;  // W: k-row bk, 8 cols at bn

  // ---- tile staging helpers ----
  auto stageA = [&](int buf, int ks) {
#if USE_ASYNC_LDS
    #pragma unroll
    for (int cc = 0; cc < 4; ++cc) {
      int c  = tid + cc * 256;                  // 1024 x 16B chunks
      int rr = c >> 3, kk = (c & 7) * 4;
      const float* src = A + (size_t)(m0 + rr) * K + ks + kk;
      __builtin_amdgcn_global_load_async_to_lds_b128(
          (__attribute__((address_space(1))) v4i_g*)src,
          (__attribute__((address_space(3))) v4i_g*)&As[buf][rr * BK + kk], 0, 0);
    }
#else
    const float* src = A + (size_t)(m0 + ar) * K + ks + ak;
    #pragma unroll
    for (int q = 0; q < 4; ++q)
      *(float4*)&As[buf][ar * BK + ak + 4 * q] = *(const float4*)(src + 4 * q);
#endif
  };
  auto stageB = [&](int buf, int ks) {
    const float* wrow = W + (size_t)(ks + bk) * N + n0 + bn;
    bf16_t* dst = &Bs[buf][bn * BK + bk];       // stride BK per column
    if (fullN) {                                 // unguarded fast path
      float4 w0 = *(const float4*)wrow;
      float4 w1 = *(const float4*)(wrow + 4);
      dst[0 * BK] = (bf16_t)w0.x; dst[1 * BK] = (bf16_t)w0.y;
      dst[2 * BK] = (bf16_t)w0.z; dst[3 * BK] = (bf16_t)w0.w;
      dst[4 * BK] = (bf16_t)w1.x; dst[5 * BK] = (bf16_t)w1.y;
      dst[6 * BK] = (bf16_t)w1.z; dst[7 * BK] = (bf16_t)w1.w;
    } else {                                     // fc3 tail blocks only
      #pragma unroll
      for (int j = 0; j < 8; ++j)
        dst[j * BK] = (bf16_t)((n0 + bn + j < N) ? wrow[j] : 0.0f);
    }
  };

  // ---- prologue: stage tile 0 ----
  stageA(0, 0);
  stageB(0, 0);
#if USE_ASYNC_LDS
  __builtin_amdgcn_s_wait_asynccnt(0);
#endif
  __syncthreads();

  int buf = 0;
  for (int ks = 0; ks < K; ks += BK) {
    // ---- prefetch next tile into the other buffer ----
    int nks = ks + BK;
    if (nks < K) { stageA(buf ^ 1, nks); stageB(buf ^ 1, nks); }

    // ---- A fragment (16x32 bf16, contiguous-k layout, same as B) ----
    // lanes 0-15: K 0..15 ; lanes 16-31: K 16..31
    const float* arow = &As[buf][(wr * 16 + r16) * BK];
    float4 a0 = *(const float4*)(arow + half * 16);
    float4 a1 = *(const float4*)(arow + half * 16 + 4);
    float4 a2 = *(const float4*)(arow + half * 16 + 8);
    float4 a3 = *(const float4*)(arow + half * 16 + 12);
    v16bf av = pack16(a0, a1, a2, a3);

    // ---- B fragments (32x16 bf16): conversion-free 32B LDS loads
    // lanes 0-15 hold K 0..15 of col r16; lanes 16-31 hold K 16..31
    #pragma unroll
    for (int ct = 0; ct < 4; ++ct) {
      const bf16_t* bcol = &Bs[buf][(ct * 16 + r16) * BK + half * 16];
      v16bf bv = *(const v16bf*)bcol;
      acc[ct] = __builtin_amdgcn_wmma_f32_16x16x32_bf16(
          false, av, false, bv, (short)0, acc[ct], false, false);
    }

#if USE_ASYNC_LDS
    __builtin_amdgcn_s_wait_asynccnt(0);   // next-tile DMA complete (no-op on last)
#endif
    __syncthreads();                        // one barrier per iteration
    buf ^= 1;
  }

  // ---- epilogue: bias, activation, scaled residual ----
  #pragma unroll
  for (int ct = 0; ct < 4; ++ct) {
    int col = n0 + ct * 16 + r16;
    if (col < N) {
      float bsv = bias ? bias[col] : 0.0f;
      #pragma unroll
      for (int v = 0; v < 8; ++v) {
        int row = m0 + wr * 16 + half * 8 + v;
        float val = acc[ct][v] + bsv;
        if (act == 1)      val = 0.5f * val * (1.0f + erff(val * 0.70710678118654752f));
        else if (act == 2) val = (val > 0.0f) ? val : 0.2f * val;
        if (R) val += resScale * R[(size_t)row * N + col];
        C[(size_t)row * N + col] = val;
      }
    }
  }
}

// ---------------------------------------------------------------------------
// 3) LayerNorm over HID=2048 per token row
// ---------------------------------------------------------------------------
__global__ __launch_bounds__(256) void ln_kernel(
    const float* __restrict__ x, const float* __restrict__ g,
    const float* __restrict__ b, float* __restrict__ y) {
  int row = blockIdx.x, tid = threadIdx.x;
  const float* xr = x + (size_t)row * HID;
  float v[8], s = 0.f, s2 = 0.f;
  #pragma unroll
  for (int e = 0; e < 8; ++e) {
    float t = xr[tid + e * 256];
    v[e] = t; s += t; s2 += t * t;
  }
  __shared__ float rs[256], rq[256];
  rs[tid] = s; rq[tid] = s2;
  __syncthreads();
  for (int off = 128; off > 0; off >>= 1) {
    if (tid < off) { rs[tid] += rs[tid + off]; rq[tid] += rq[tid + off]; }
    __syncthreads();
  }
  float mean = rs[0] * (1.0f / HID);
  float var  = rq[0] * (1.0f / HID) - mean * mean;
  float inv  = rsqrtf(var + 1e-5f);
  #pragma unroll
  for (int e = 0; e < 8; ++e) {
    int c = tid + e * 256;
    y[(size_t)row * HID + c] = (v[e] - mean) * inv * g[c] + b[c];
  }
}

// ---------------------------------------------------------------------------
// 4) Attention core: SEQ=4, DH=512. One block per (batch, head).
// ---------------------------------------------------------------------------
__global__ __launch_bounds__(256) void attn_kernel(
    const float* __restrict__ qkv, float* __restrict__ o) {
  int b = blockIdx.x >> 2, hd = blockIdx.x & 3;
  int tid = threadIdx.x;
  const float* base = qkv + (size_t)b * 4 * 6144 + hd * 512;

  __shared__ float red[16 * 256];
  __shared__ float scores[16];
  __shared__ float probs[16];

  float part[16];
  #pragma unroll
  for (int p = 0; p < 16; ++p) part[p] = 0.f;

  for (int d = tid; d < 512; d += 256) {
    float qv[4], kv[4];
    #pragma unroll
    for (int i = 0; i < 4; ++i) qv[i] = base[(size_t)i * 6144 + d];
    #pragma unroll
    for (int j = 0; j < 4; ++j) kv[j] = base[(size_t)j * 6144 + 2048 + d];
    #pragma unroll
    for (int i = 0; i < 4; ++i)
      #pragma unroll
      for (int j = 0; j < 4; ++j) part[i * 4 + j] += qv[i] * kv[j];
  }
  #pragma unroll
  for (int p = 0; p < 16; ++p) red[p * 256 + tid] = part[p];
  __syncthreads();
  if (tid < 16) {
    float s = 0.f;
    for (int t = 0; t < 256; ++t) s += red[tid * 256 + t];
    scores[tid] = s * ATTN_SCALE;
  }
  __syncthreads();
  if (tid < 4) {
    float sv[4], mx = -1e30f, sum = 0.f;
    #pragma unroll
    for (int j = 0; j < 4; ++j) { sv[j] = scores[tid * 4 + j]; mx = fmaxf(mx, sv[j]); }
    #pragma unroll
    for (int j = 0; j < 4; ++j) { sv[j] = __expf(sv[j] - mx); sum += sv[j]; }
    float r = 1.0f / sum;
    #pragma unroll
    for (int j = 0; j < 4; ++j) probs[tid * 4 + j] = sv[j] * r;
  }
  __syncthreads();
  for (int d = tid; d < 512; d += 256) {
    float vv[4];
    #pragma unroll
    for (int j = 0; j < 4; ++j) vv[j] = base[(size_t)j * 6144 + 4096 + d];
    #pragma unroll
    for (int i = 0; i < 4; ++i) {
      float ov = 0.f;
      #pragma unroll
      for (int j = 0; j < 4; ++j) ov += probs[i * 4 + j] * vv[j];
      o[((size_t)b * 4 + i) * HID + hd * 512 + d] = ov;
    }
  }
}

// ---------------------------------------------------------------------------
// Host orchestration
// ---------------------------------------------------------------------------
extern "C" void kernel_launch(void* const* d_in, const int* in_sizes, int n_in,
                              void* d_out, int out_size, void* d_ws, size_t ws_size,
                              hipStream_t stream) {
  const float* x       = (const float*)d_in[0];
  const float* mask    = (const float*)d_in[1];
  const float* embed_w = (const float*)d_in[2];
  const float* embed_b = (const float*)d_in[3];
  const float* ln_g    = (const float*)d_in[4];
  const float* ln_b    = (const float*)d_in[5];
  const float* mlp_w1  = (const float*)d_in[6];
  const float* mlp_b1  = (const float*)d_in[7];
  const float* mlp_w2  = (const float*)d_in[8];
  const float* mlp_b2  = (const float*)d_in[9];
  const float* qkv_w   = (const float*)d_in[10];
  const float* qkv_b   = (const float*)d_in[11];
  const float* out_w   = (const float*)d_in[12];
  const float* out_b   = (const float*)d_in[13];
  const float* fc1_w   = (const float*)d_in[14];
  const float* fc1_b   = (const float*)d_in[15];
  const float* fc2_w   = (const float*)d_in[16];
  const float* fc2_b   = (const float*)d_in[17];
  const float* fc3_w   = (const float*)d_in[18];
  const float* fc3_b   = (const float*)d_in[19];
  float* out = (float*)d_out;

  // Workspace carve-out (~80 MB of fp32)
  float* t    = (float*)d_ws;                     // 1024 x 3072
  float* h    = t    + (size_t)ROWS * EMBED_IN;   // 1024 x 2048
  float* y    = h    + (size_t)ROWS * HID;        // 1024 x 2048
  float* mmid = y    + (size_t)ROWS * HID;        // 1024 x 4096
  float* qkvb = mmid + (size_t)ROWS * 4096;       // 1024 x 6144
  float* ob   = qkvb + (size_t)ROWS * 6144;       // 1024 x 2048

  auto gemm = [&](const float* A, const float* W, const float* bias,
                  const float* R, float rs, float* Cp,
                  int M, int N, int K, int act) {
    dim3 grid((N + BN - 1) / BN, M / BM);
    gemm_kernel<<<grid, 256, 0, stream>>>(A, W, bias, R, rs, Cp, M, N, K, act);
  };

  // 1) patchify + mask
  patch_kernel<<<(BATCH * 12288 + 255) / 256, 256, 0, stream>>>(x, mask, t);

  // 2) embed: h = t @ embed_w + embed_b
  gemm(t, embed_w, embed_b, nullptr, 0.f, h, ROWS, HID, EMBED_IN, 0);

  // 3) MLP blocks: h += gelu(LN(h) @ W1 + b1) @ W2 + b2
  for (int i = 0; i < 4; ++i) {
    ln_kernel<<<ROWS, 256, 0, stream>>>(h, ln_g + (size_t)i * HID,
                                        ln_b + (size_t)i * HID, y);
    gemm(y, mlp_w1 + (size_t)i * HID * 4096, mlp_b1 + (size_t)i * 4096,
         nullptr, 0.f, mmid, ROWS, 4096, HID, 1);
    gemm(mmid, mlp_w2 + (size_t)i * 4096 * HID, mlp_b2 + (size_t)i * HID,
         h, 1.0f, h, ROWS, HID, 4096, 0);
  }

  // 4) Attention blocks: h = 2*h + softmax(qk)v @ out_w + out_b
  for (int i = 0; i < 4; ++i) {
    gemm(h, qkv_w + (size_t)i * HID * 6144, qkv_b + (size_t)i * 6144,
         nullptr, 0.f, qkvb, ROWS, 6144, HID, 0);
    attn_kernel<<<BATCH * 4, 256, 0, stream>>>(qkvb, ob);
    gemm(ob, out_w + (size_t)i * HID * HID, out_b + (size_t)i * HID,
         h, 2.0f, h, ROWS, HID, HID, 0);
  }

  // 5) Head: f = h.reshape(256, 8192); two leaky-relu FCs, final classifier
  gemm(h, fc1_w, fc1_b, nullptr, 0.f, y, BATCH, HID, SEQ * HID, 2);   // 256x2048
  gemm(y, fc2_w, fc2_b, nullptr, 0.f, mmid, BATCH, HID, HID, 2);      // 256x2048
  gemm(mmid, fc3_w, fc3_b, nullptr, 0.f, out, BATCH, NCLASS, HID, 0); // 256x1000
}
